// MLP_43293270343765
// MI455X (gfx1250) — matrix operand order repaired and checked by reading, not compiled, not run on hardware
//
#include <hip/hip_runtime.h>

typedef __attribute__((ext_vector_type(16))) _Float16 v16h;
typedef __attribute__((ext_vector_type(8)))  _Float16 v8h;
typedef __attribute__((ext_vector_type(4)))  _Float16 v4h;
typedef __attribute__((ext_vector_type(8)))  float    v8f;
typedef __attribute__((ext_vector_type(4)))  float    v4f;

namespace {
constexpr int NROWS  = 500000;
constexpr int KCTX   = 20;
constexpr int GRP    = 25;          // nr + 1
constexpr int H      = 128;
constexpr int F      = 384;         // 3*H
constexpr int H2     = 64;
constexpr int IDXS   = 23;          // 3 + K
constexpr int NTILES = NROWS / 16;  // 31250 (exact)
constexpr int WAVES  = 8;
constexpr int BLOCK  = 32 * WAVES;
constexpr int GRIDX  = 977;         // ~4 tiles per wave

// LDS strides in element units (all keep 16B alignment for b128 ds ops)
constexpr int LD1  = 392;   // W1h row stride (halfs)
constexpr int LD2  = 136;   // W2h row stride (halfs)
constexpr int LDA  = 392;   // feature tile stride (halfs)
constexpr int LDY1 = 136;   // Y1 stride (halfs)
constexpr int LDY2 = 72;    // Y2 stride (floats)

constexpr size_t O_W1 = 0;
constexpr size_t O_W2 = O_W1 + size_t(H)  * LD1 * 2;
constexpr size_t O_B1 = O_W2 + size_t(H2) * LD2 * 2;
constexpr size_t O_B2 = O_B1 + size_t(H)  * 4;
constexpr size_t O_W3 = O_B2 + size_t(H2) * 4;
constexpr size_t O_B3 = O_W3 + size_t(H2) * 4;
constexpr size_t O_PW = ((O_B3 + 4 + 127) / 128) * 128;
constexpr size_t PW_A  = 0;
constexpr size_t PW_Y1 = PW_A  + size_t(16) * LDA  * 2;
constexpr size_t PW_Y2 = PW_Y1 + size_t(16) * LDY1 * 2;
constexpr size_t PW_SZ = ((PW_Y2 + size_t(16) * LDY2 * 4 + 127) / 128) * 128;
constexpr size_t SMEM_BYTES = O_PW + PW_SZ * WAVES;   // ~284 KB (<320 KB/WG)
} // namespace

__device__ __forceinline__ v16h cat8(v8h lo, v8h hi) {
  v16h r;
#pragma unroll
  for (int i = 0; i < 8; ++i) { r[i] = lo[i]; r[i + 8] = hi[i]; }
  return r;
}

// A fragment (16x32 f16) from row-major LDS tile, per ISA 16-bit A layout:
// lane<16: row=lane, K = [k0..k0+7] U [k0+16..k0+23]
// lane>=16: row=lane-16, K = [k0+8..k0+15] U [k0+24..k0+31]
__device__ __forceinline__ v16h ldA(const _Float16* tile, int lane, int k0, int ld) {
  const int m = lane & 15, h = lane >> 4;
  const _Float16* p = tile + m * ld + k0 + h * 8;
  v8h lo = *(const v8h*)(p);
  v8h hi = *(const v8h*)(p + 16);
  return cat8(lo, hi);
}

// B fragment (32x16 f16) from row-major weight [n][k] (i.e. B = Wt):
// lane<16: col=lane, K = [k0..k0+7] U [k0+16..k0+23] ; lane>=16: col=lane-16, K = [k0+8..k0+15] U [k0+24..k0+31]
__device__ __forceinline__ v16h ldB(const _Float16* w, int lane, int n0, int k0, int ld) {
  const int n = lane & 15, h = lane >> 4;
  const _Float16* p = w + (n0 + n) * ld + k0 + h * 8;
  v8h lo = *(const v8h*)(p);
  v8h hi = *(const v8h*)(p + 16);
  return cat8(lo, hi);
}

__device__ __forceinline__ v4f group_mean(const int* __restrict__ idx,
                                          const float* __restrict__ x,
                                          int g, int lane) {
  const int* cb = idx + g * GRP * IDXS + 3;
  v4f acc = {0.f, 0.f, 0.f, 0.f};
  int cnt = 0;
#pragma unroll 5
  for (int j = 0; j < KCTX; ++j) {
    const int node = cb[j];
    cnt += (node > 0) ? 1 : 0;
    v4f e = *(const v4f*)(x + size_t(node) * H + 4 * lane);
    acc += e;
  }
  const float inv = 1.0f / (float)(cnt < 1 ? 1 : cnt);
  return acc * inv;
}

__global__ __launch_bounds__(BLOCK)
void MLP_43293270343765_kernel(const int* __restrict__ idx,
                               const float* __restrict__ x,
                               const float* __restrict__ W1, const float* __restrict__ b1,
                               const float* __restrict__ W2, const float* __restrict__ b2,
                               const float* __restrict__ W3, const float* __restrict__ b3,
                               float* __restrict__ out) {
  extern __shared__ char smem[];
  _Float16* W1h = (_Float16*)(smem + O_W1);
  _Float16* W2h = (_Float16*)(smem + O_W2);
  float* b1f = (float*)(smem + O_B1);
  float* b2f = (float*)(smem + O_B2);
  float* w3f = (float*)(smem + O_W3);
  float* b3f = (float*)(smem + O_B3);

  const int tid = threadIdx.x;

  // ---- stage weights to f16 LDS (block cooperative) ----
  for (int i = tid; i < H * F; i += BLOCK) {
    const int r = i / F, c = i - r * F;
    W1h[r * LD1 + c] = (_Float16)W1[i];
  }
  for (int i = tid; i < H2 * H; i += BLOCK) {
    const int r = i >> 7, c = i & 127;
    W2h[r * LD2 + c] = (_Float16)W2[i];
  }
  if (tid < H)  b1f[tid] = b1[tid];
  if (tid < H2) b2f[tid] = b2[tid];
  if (tid < H2) w3f[tid] = W3[tid];
  if (tid == 0) b3f[0] = b3[0];
  __syncthreads();

  const int lane = tid & 31;
  const int wv   = tid >> 5;
  const int nlo  = lane & 15;
  const int mhi  = (lane >> 4) * 8;

  char* pw = smem + O_PW + size_t(wv) * PW_SZ;
  _Float16* At = (_Float16*)(pw + PW_A);
  _Float16* Y1 = (_Float16*)(pw + PW_Y1);
  float*    Y2 = (float*)(pw + PW_Y2);

  const float b3s = b3f[0];

  for (int t = blockIdx.x * WAVES + wv; t < NTILES; t += GRIDX * WAVES) {
    const int r0 = t * 16;

    // ---- build 16x384 f16 feature tile: [x[i0] | x[i1] | group mean] ----
#pragma unroll 4
    for (int m = 0; m < 16; ++m) {
      const int row = r0 + m;
      const int i0 = idx[row * IDXS + 0];
      const int i1 = idx[row * IDXS + 1];
      v4f e0 = *(const v4f*)(x + size_t(i0) * H + 4 * lane);
      v4f e1 = *(const v4f*)(x + size_t(i1) * H + 4 * lane);
      v4h h0, h1;
#pragma unroll
      for (int j = 0; j < 4; ++j) { h0[j] = (_Float16)e0[j]; h1[j] = (_Float16)e1[j]; }
      *(v4h*)(At + m * LDA + 4 * lane)     = h0;
      *(v4h*)(At + m * LDA + H + 4 * lane) = h1;
    }
    const int g0 = r0 / GRP;
    const int g1 = (r0 + 15) / GRP;
    v4f mean0 = group_mean(idx, x, g0, lane);
    v4f mean1 = (g1 != g0) ? group_mean(idx, x, g1, lane) : mean0;
#pragma unroll 4
    for (int m = 0; m < 16; ++m) {
      const v4f mv = ((r0 + m) / GRP == g0) ? mean0 : mean1;
      v4h hm;
#pragma unroll
      for (int j = 0; j < 4; ++j) hm[j] = (_Float16)mv[j];
      *(v4h*)(At + m * LDA + 2 * H + 4 * lane) = hm;
    }
    __builtin_amdgcn_wave_barrier();   // wave-synchronous LDS fill -> fragment loads

    // ---- layer 1: (16x384) @ (384x128) via 96 WMMAs ----
    v16h a[12];
#pragma unroll
    for (int kt = 0; kt < 12; ++kt) a[kt] = ldA(At, lane, kt * 32, LDA);
#pragma unroll
    for (int nt = 0; nt < 8; ++nt) {
      const float bias = b1f[nt * 16 + nlo];
      v8f c;
#pragma unroll
      for (int i = 0; i < 8; ++i) c[i] = bias;
#pragma unroll
      for (int kt = 0; kt < 12; ++kt) {
        v16h b = ldB(W1h, lane, nt * 16, kt * 32, LD1);
        c = __builtin_amdgcn_wmma_f32_16x16x32_f16(false, a[kt], false, b,
                                                   (short)0, c, false, false);
      }
#pragma unroll
      for (int i = 0; i < 8; ++i) {
        const float vv = fmaxf(c[i], 0.0f);
        Y1[(i + mhi) * LDY1 + nt * 16 + nlo] = (_Float16)vv;   // relu -> f16
      }
    }
    __builtin_amdgcn_wave_barrier();

    // ---- layer 2: (16x128) @ (128x64) via 16 WMMAs ----
    v16h a2[4];
#pragma unroll
    for (int kt = 0; kt < 4; ++kt) a2[kt] = ldA(Y1, lane, kt * 32, LDY1);
#pragma unroll
    for (int nt = 0; nt < 4; ++nt) {
      const float bias = b2f[nt * 16 + nlo];
      v8f c;
#pragma unroll
      for (int i = 0; i < 8; ++i) c[i] = bias;
#pragma unroll
      for (int kt = 0; kt < 4; ++kt) {
        v16h b = ldB(W2h, lane, nt * 16, kt * 32, LD2);
        c = __builtin_amdgcn_wmma_f32_16x16x32_f16(false, a2[kt], false, b,
                                                   (short)0, c, false, false);
      }
#pragma unroll
      for (int i = 0; i < 8; ++i)
        Y2[(i + mhi) * LDY2 + nt * 16 + nlo] = fmaxf(c[i], 0.0f);  // relu, f32
    }
    __builtin_amdgcn_wave_barrier();

    // ---- layer 3: 64-wide dot per row (VALU), split across lane halves ----
    {
      const int m = lane & 15;
      const int hf = lane >> 4;
      const float* yr = Y2 + m * LDY2 + hf * 32;
      const float* wr = w3f + hf * 32;
      float s = 0.0f;
#pragma unroll
      for (int j = 0; j < 32; ++j) s += yr[j] * wr[j];
      s += __shfl_xor(s, 16, 32);
      if (hf == 0) out[r0 + m] = s + b3s;
    }
    __builtin_amdgcn_wave_barrier();
  }
}

extern "C" void kernel_launch(void* const* d_in, const int* in_sizes, int n_in,
                              void* d_out, int out_size, void* d_ws, size_t ws_size,
                              hipStream_t stream) {
  const int*   idx = (const int*)  d_in[0];
  // d_in[1] = nr (24), compile-time constant here
  const float* x   = (const float*)d_in[2];
  const float* W1  = (const float*)d_in[3];
  const float* b1  = (const float*)d_in[4];
  const float* W2  = (const float*)d_in[5];
  const float* b2  = (const float*)d_in[6];
  const float* W3  = (const float*)d_in[7];
  const float* b3  = (const float*)d_in[8];
  float* out = (float*)d_out;

  (void)in_sizes; (void)n_in; (void)out_size; (void)d_ws; (void)ws_size;

  // CDNA5 allows up to 320KB LDS per workgroup; raise the dynamic-LDS cap.
  hipFuncSetAttribute((const void*)MLP_43293270343765_kernel,
                      hipFuncAttributeMaxDynamicSharedMemorySize,
                      (int)SMEM_BYTES);

  MLP_43293270343765_kernel<<<GRIDX, BLOCK, SMEM_BYTES, stream>>>(
      idx, x, W1, b1, W2, b2, W3, b3, out);
}
